// TGAN_71528385347818
// MI455X (gfx1250) — matrix operand order repaired and checked by reading, not compiled, not run on hardware
//
#include <hip/hip_runtime.h>
#include <hip/hip_bf16.h>
#include <math.h>

// ---------------------------------------------------------------------------
// TGAN graph-transformer forward for MI455X (gfx1250, wave32, WMMA).
// B=64, N=64, D=128, NH=2, L=2. All GEMMs via v_wmma_f32_16x16x32_bf16 with
// LDS operands pre-swizzled into fragment layout (32B contiguous per lane ->
// 2x ds_load_b128 per fragment). Staging uses packed v2bf/v4bf LDS stores.
// Edge-bias GEMM is fused into the attention kernel (no [B,N,N,D] bias in HBM).
// ---------------------------------------------------------------------------

typedef __bf16 bf16;
typedef __attribute__((ext_vector_type(2)))  __bf16 v2bf;
typedef __attribute__((ext_vector_type(4)))  __bf16 v4bf;
typedef __attribute__((ext_vector_type(16))) __bf16 v16bf;
typedef __attribute__((ext_vector_type(8)))  float  v8f;

static constexpr int Bn = 64, Nn = 64, NHn = 2;
static constexpr float INV_TEMP = 0.0883883476483184f;   // 1/sqrt(128)

// ---- fragment-layout swizzles (CDNA5 ISA 7.12.2, wave32) -------------------
// One 16x16x32 operand tile = 32 lanes x 16 bf16 = 512 halves = 1 KB,
// stored as tile[lane*16 + e].

// A element (m in 0..15, k in 0..31): consecutive (4-aligned) k -> consecutive e
__device__ __forceinline__ int swA(int m, int k) {
  return (m + ((k >> 3) & 1) * 16) * 16 + ((k & 7) | ((k & 16) >> 1));
}
// B element (k in 0..31, n in 0..15): consecutive (2-aligned) k -> consecutive e
__device__ __forceinline__ int swB(int k, int n) {
  return (n + ((k >> 3) & 1) * 16) * 16 + ((k & 7) | ((k & 16) >> 1));
}
// Fragment load: 32 contiguous bytes per lane (2x ds_load_b128)
__device__ __forceinline__ v16bf frag_ld(const bf16* tile) {
  return *(const v16bf*)(tile + (threadIdx.x & 31) * 16);
}
// Packed bf16 stores (ds_store_b32 / ds_store_b64)
__device__ __forceinline__ void st2(bf16* dst, float x, float y) {
  v2bf t; t[0] = (bf16)x; t[1] = (bf16)y;
  *(v2bf*)dst = t;
}
__device__ __forceinline__ void st4(bf16* dst, float4 v) {
  v4bf t; t[0] = (bf16)v.x; t[1] = (bf16)v.y; t[2] = (bf16)v.z; t[3] = (bf16)v.w;
  *(v4bf*)dst = t;
}

__device__ __forceinline__ v8f wmma_bf16(v8f acc, v16bf a, v16bf b) {
  return __builtin_amdgcn_wmma_f32_16x16x32_bf16(
      false, a, false, b, (short)0, acc, false, false);
}

// ---------------------------------------------------------------------------
// Kernel 1: emb = [node_feat || stpe_emb][4096,256] @ Wn[256,128] + bn
// ---------------------------------------------------------------------------
__global__ void node_proj_kernel(const float* __restrict__ node_feat,
                                 const float* __restrict__ stpe,
                                 const float* __restrict__ Wn,
                                 const float* __restrict__ bn,
                                 float* __restrict__ emb) {
  __shared__ bf16 Ab[4 * 512];    // 4 m-tiles, swizzled
  __shared__ bf16 Bb[8 * 512];    // 8 n-tiles, swizzled
  const int tid  = threadIdx.x;
  const int row0 = blockIdx.x * 64;
  const int w    = tid >> 5;
  const int mt   = w & 3;
  const int nt0  = (w >> 2) * 4;
  const int lane = tid & 31;
  const int colw = lane & 15;
  const int rb   = (lane < 16) ? 0 : 8;

  v8f acc[4] = {};
  for (int k0 = 0; k0 < 256; k0 += 32) {
    for (int idx = tid; idx < 512; idx += 256) {       // A: float4 -> v4bf
      const int r = idx >> 3, c4 = (idx & 7) * 4;
      const int gk = k0 + c4;
      const float4 v4 = (gk < 128)
          ? *(const float4*)(node_feat + (row0 + r) * 128 + gk)
          : *(const float4*)(stpe + (row0 + r) * 128 + (gk - 128));
      st4(Ab + (r >> 4) * 512 + swA(r & 15, c4), v4);
    }
    for (int idx = tid; idx < 2048; idx += 256) {      // B: 2 k-rows -> v2bf
      const int k2 = (idx >> 7) * 2, c = idx & 127;
      st2(Bb + (c >> 4) * 512 + swB(k2, c & 15),
          Wn[(k0 + k2) * 128 + c], Wn[(k0 + k2 + 1) * 128 + c]);
    }
    __syncthreads();
    const v16bf a = frag_ld(Ab + mt * 512);
#pragma unroll
    for (int j = 0; j < 4; ++j)
      acc[j] = wmma_bf16(acc[j], a, frag_ld(Bb + (nt0 + j) * 512));
    __syncthreads();
  }
#pragma unroll
  for (int j = 0; j < 4; ++j) {
    const int col = (nt0 + j) * 16 + colw;
#pragma unroll
    for (int r = 0; r < 8; ++r)
      emb[(row0 + mt * 16 + rb + r) * 128 + col] = acc[j][r] + bn[col];
  }
}

// ---------------------------------------------------------------------------
// Kernel 2 (per layer): h = LN(emb); {q,k,v} = h @ W{q,k,v}[128,256]
// grid 64*3 (batch x matrix). Output layout [h][b][n][128] fp32.
// ---------------------------------------------------------------------------
__global__ void qkv_kernel(const float* __restrict__ emb,
                           const float* __restrict__ ln_g,
                           const float* __restrict__ ln_b,
                           const float* __restrict__ Wq,
                           const float* __restrict__ Wk,
                           const float* __restrict__ Wv,
                           float* __restrict__ qo,
                           float* __restrict__ ko,
                           float* __restrict__ vo) {
  __shared__ bf16 Hb[4 * 4 * 512];   // [mtile][kblk] swizzled A
  __shared__ bf16 Wb[16 * 512];      // [ntile] swizzled B (one 32-K block)
  const int tid = threadIdx.x;
  const int b   = blockIdx.x / 3;
  const int mat = blockIdx.x % 3;
  const float* W = (mat == 0) ? Wq : (mat == 1) ? Wk : Wv;
  float*     out = (mat == 0) ? qo : (mat == 1) ? ko : vo;

  if (tid < 64) {                                  // pre-LN (eps 1e-5)
    const float* row = emb + (b * 64 + tid) * 128;
    float mu = 0.f;
    for (int c = 0; c < 128; ++c) mu += row[c];
    mu *= (1.f / 128.f);
    float var = 0.f;
    for (int c = 0; c < 128; ++c) { const float d = row[c] - mu; var += d * d; }
    var *= (1.f / 128.f);
    const float rstd = rsqrtf(var + 1e-5f);
    const int mt = tid >> 4, m = tid & 15;
    for (int c = 0; c < 128; c += 2)
      st2(Hb + (mt * 4 + (c >> 5)) * 512 + swA(m, c & 31),
          (row[c] - mu) * rstd * ln_g[c] + ln_b[c],
          (row[c + 1] - mu) * rstd * ln_g[c + 1] + ln_b[c + 1]);
  }
  __syncthreads();

  const int w    = tid >> 5;
  const int mt   = w & 3;
  const int nt0  = (w >> 2) * 8;
  const int lane = tid & 31;
  const int colw = lane & 15;
  const int rb   = (lane < 16) ? 0 : 8;

  v8f acc[8] = {};
  for (int kb = 0; kb < 4; ++kb) {
    for (int idx = tid; idx < 4096; idx += 256) {      // 2 k-rows -> v2bf
      const int k2 = (idx >> 8) * 2, c = idx & 255;
      st2(Wb + (c >> 4) * 512 + swB(k2, c & 15),
          W[(kb * 32 + k2) * 256 + c], W[(kb * 32 + k2 + 1) * 256 + c]);
    }
    __syncthreads();
    const v16bf a = frag_ld(Hb + (mt * 4 + kb) * 512);
#pragma unroll
    for (int j = 0; j < 8; ++j)
      acc[j] = wmma_bf16(acc[j], a, frag_ld(Wb + (nt0 + j) * 512));
    __syncthreads();
  }
#pragma unroll
  for (int j = 0; j < 8; ++j) {
    const int c  = (nt0 + j) * 16 + colw;
    const int h  = c >> 7, dk = c & 127;
#pragma unroll
    for (int r = 0; r < 8; ++r) {
      const int n = mt * 16 + rb + r;
      out[(((h * 64 + b) * 64) + n) * 128 + dk] = acc[j][r];
    }
  }
}

// ---------------------------------------------------------------------------
// Kernel 3 (per layer): fused edge-bias + attention + out-proj + FFN.
// grid = B*4 (batch x 16-row i-block) x 256thr. ~169 KB dynamic LDS arena
// with phase-based region reuse (320 KB/WGP available on CDNA5).
// ---------------------------------------------------------------------------
#define ATTN_SMEM_BYTES 173056

__global__ void attn_kernel(const float* __restrict__ edge,
                            const unsigned char* __restrict__ mask,  // jnp bool = 1B
                            const float* __restrict__ We, const float* __restrict__ be,
                            const float* __restrict__ qws, const float* __restrict__ kws,
                            const float* __restrict__ vws,
                            const float* __restrict__ Wo, const float* __restrict__ bo,
                            const float* __restrict__ ln2g, const float* __restrict__ ln2b,
                            const float* __restrict__ W1, const float* __restrict__ b1,
                            const float* __restrict__ W2, const float* __restrict__ b2,
                            float* __restrict__ embio,
                            float* __restrict__ attn_out) {
  extern __shared__ char smem[];
  bf16*  WeS  = (bf16*)(smem + 0);        // [4 kblk][8 nt][512]; FFN: W1 stage [32 nt][512]
  bf16*  W1S  = (bf16*)(smem + 0);
  float* qF   = (float*)(smem + 32768);   // q fp32 [2][16][128]
  bf16*  qS   = (bf16*)(smem + 49152);    // q A-frags [2][4 kblk][512]; later outcat [8][512]
  bf16*  ocS  = (bf16*)(smem + 49152);
  bf16*  kS   = (bf16*)(smem + 57344);    // k B^T-frags [2][4 kblk][4 jt][512]; later T fp32
  float* Tf   = (float*)(smem + 57344);   // [64][128]
  bf16*  vS   = (bf16*)(smem + 90112);    // v B-frags [2][2 kblk][8 nt][512]
  float* a1   = (float*)(smem + 122880);  // attn1 [2][16][64]; later Wo/W2 stage [8][512]
  bf16*  wstg = (bf16*)(smem + 122880);
  bf16*  eg   = (bf16*)(smem + 131072);   // edge A-frags [4 mt][512]; later hn [4 kblk][512]
  bf16*  hnS  = (bf16*)(smem + 131072);
  float* red  = (float*)(smem + 135680);  // [8]: per-wave max / expsum partials
  float* aF   = (float*)(smem + 136192);  // attn fp32 [2][16][64]
  bf16*  aS   = (bf16*)(smem + 144384);   // attn A-frags [2][2 kblk][512]
  float* e12  = (float*)(smem + 148480);  // emb1+emb2 [2][16][128]; later g [16 kblk][512]
  bf16*  gS   = (bf16*)(smem + 148480);
  float* ne   = (float*)(smem + 164864);  // newemb [16][128]

  const int tid  = threadIdx.x;
  const int lane = tid & 31;
  const int w    = tid >> 5;
  const int b    = blockIdx.x >> 2;
  const int i0   = (blockIdx.x & 3) * 16;
  const int colw = lane & 15;
  const int rb   = (lane < 16) ? 0 : 8;

  // ---- Phase 1: stage We, q, k, v (all pre-swizzled, packed stores) ----
  for (int idx = tid; idx < 8192; idx += 256) {        // We [128][128]
    const int k2 = (idx >> 7) * 2, d = idx & 127;
    st2(WeS + ((k2 >> 5) * 8 + (d >> 4)) * 512 + swB(k2 & 31, d & 15),
        We[k2 * 128 + d], We[(k2 + 1) * 128 + d]);
  }
  for (int idx = tid; idx < 2048; idx += 256) {        // q [2][16][128]
    const int h = idx >> 10, rem = idx & 1023;
    const int ii = rem >> 6, d2 = (rem & 63) * 2;
    const float2 v2 = *(const float2*)(qws + ((h * 64 + b) * 64 + (i0 + ii)) * 128 + d2);
    *(float2*)(qF + h * 2048 + ii * 128 + d2) = v2;
    st2(qS + (h * 4 + (d2 >> 5)) * 512 + swA(ii, d2 & 31), v2.x, v2.y);
  }
  for (int idx = tid; idx < 8192; idx += 256) {        // k: B^T frags (K=d, N=j)
    const int h = idx >> 12, rem = idx & 4095;
    const int j = rem >> 6, d2 = (rem & 63) * 2;
    const float2 v2 = *(const float2*)(kws + ((size_t)(h * 64 + b) * 64 + j) * 128 + d2);
    st2(kS + ((h * 4 + (d2 >> 5)) * 4 + (j >> 4)) * 512 + swB(d2 & 31, j & 15), v2.x, v2.y);
  }
  for (int idx = tid; idx < 8192; idx += 256) {        // v: B frags (K=j, N=d)
    const int h = idx >> 12, rem = idx & 4095;
    const int j2 = (rem >> 7) * 2, d = rem & 127;
    const float* vp = vws + ((size_t)(h * 64 + b) * 64 + j2) * 128 + d;
    st2(vS + ((h * 2 + (j2 >> 5)) * 8 + (d >> 4)) * 512 + swB(j2 & 31, d & 15),
        vp[0], vp[128]);
  }
  __syncthreads();

  // ---- attn1 = q @ k^T for the 16-row block (per head) ----
  {
    const int h = w >> 2, nt = w & 3;
    v8f acc = {};
#pragma unroll
    for (int kb = 0; kb < 4; ++kb)
      acc = wmma_bf16(acc, frag_ld(qS + (h * 4 + kb) * 512),
                      frag_ld(kS + ((h * 4 + kb) * 4 + nt) * 512));
#pragma unroll
    for (int r = 0; r < 8; ++r)
      a1[h * 1024 + (rb + r) * 64 + nt * 16 + colw] = acc[r];
  }
  __syncthreads();

  // ---- Phase 2: per query row i — T_i GEMM, scores, softmax, emb2 ----
  for (int i = 0; i < 16; ++i) {
    const int qi = i0 + i;
    const float* E = edge + (size_t)(b * 64 + qi) * 64 * 128;
    if (i + 1 < 16 && tid < 64)          // gfx1250 global_prefetch of next row block
      __builtin_prefetch(E + 64 * 128 + tid * 128, 0, 0);

    const int mt = w & 3, nt0 = (w >> 2) * 4;
    v8f accT[4] = {};
    for (int kb = 0; kb < 4; ++kb) {
      for (int idx = tid; idx < 512; idx += 256) {     // 64x32: float4 -> v4bf
        const int r = idx >> 3, c4 = (idx & 7) * 4;
        const float4 v4 = *(const float4*)(E + r * 128 + kb * 32 + c4);
        st4(eg + (r >> 4) * 512 + swA(r & 15, c4), v4);
      }
      __syncthreads();
      const v16bf a = frag_ld(eg + mt * 512);
#pragma unroll
      for (int j = 0; j < 4; ++j)
        accT[j] = wmma_bf16(accT[j], a, frag_ld(WeS + (kb * 8 + nt0 + j) * 512));
      __syncthreads();
    }
#pragma unroll
    for (int j = 0; j < 4; ++j) {                      // T = edge@We + be (fp32 LDS)
      const int col = (nt0 + j) * 16 + colw;
#pragma unroll
      for (int r = 0; r < 8; ++r)
        Tf[(mt * 16 + rb + r) * 128 + col] = accT[j][r] + be[col];
    }
    __syncthreads();

    const bool act = tid < 128;                        // wave 0,1 -> head0; 2,3 -> head1
    float s = 0.f;
    if (act) {                                         // scores = (attn1 + q.T)/temp
      const int h = tid >> 6, j = tid & 63;
      const float4* qv = (const float4*)(qF + (h * 16 + i) * 128);
      const float4* tv = (const float4*)(Tf + j * 128);
      float s2 = 0.f;
#pragma unroll 8
      for (int c = 0; c < 32; ++c) {
        const float4 a4 = qv[c], b4 = tv[c];
        s2 += a4.x * b4.x + a4.y * b4.y + a4.z * b4.z + a4.w * b4.w;
      }
      s = (a1[h * 1024 + i * 64 + j] + s2) * INV_TEMP;
      if (mask[(size_t)(b * 64 + qi) * 64 + j]) s = 1e-10f;  // faithful masked_fill
      float m = s;                                     // wave-level max
#pragma unroll
      for (int off = 16; off; off >>= 1) m = fmaxf(m, __shfl_xor(m, off, 32));
      if (lane == 0) red[w] = m;
    }
    __syncthreads();
    float ev = 0.f;
    if (act) {
      const int h = tid >> 6;
      const float m = fmaxf(red[h * 2], red[h * 2 + 1]);
      ev = __expf(s - m);
      float ss = ev;                                   // wave-level sum
#pragma unroll
      for (int off = 16; off; off >>= 1) ss += __shfl_xor(ss, off, 32);
      if (lane == 0) red[4 + w] = ss;
    }
    __syncthreads();
    if (act) {
      const int h = tid >> 6, j = tid & 63;
      const float av = ev / (red[4 + h * 2] + red[4 + h * 2 + 1]);
      aF[h * 1024 + i * 64 + j] = av;
      aS[(h * 2 + (j >> 5)) * 512 + swA(i, j & 31)] = (bf16)av;
      attn_out[(((size_t)h * 64 + b) * 64 + qi) * 64 + j] = av;
    }
    __syncthreads();
    {                                                  // emb2 = attn_row @ T_i
      const int h = tid >> 7, d = tid & 127;
      const float* ar = aF + h * 1024 + i * 64;
      float accv = 0.f;
#pragma unroll 8
      for (int j = 0; j < 64; ++j) accv += ar[j] * Tf[j * 128 + d];
      e12[h * 2048 + i * 128 + d] = accv;
    }
    __syncthreads();
  }

  // ---- Phase 3: emb1 = attn @ v (WMMA), out-proj, residual, FFN ----
  {
    const int h = w >> 2, nt0e = (w & 3) * 2;
    v8f acc[2] = {};
#pragma unroll
    for (int kb = 0; kb < 2; ++kb) {
      const v16bf a = frag_ld(aS + (h * 2 + kb) * 512);
#pragma unroll
      for (int j = 0; j < 2; ++j)
        acc[j] = wmma_bf16(acc[j], a, frag_ld(vS + ((h * 2 + kb) * 8 + nt0e + j) * 512));
    }
#pragma unroll
    for (int j = 0; j < 2; ++j) {
      const int col = (nt0e + j) * 16 + colw;
#pragma unroll
      for (int r = 0; r < 8; ++r)
        e12[h * 2048 + (rb + r) * 128 + col] += acc[j][r];
    }
  }
  __syncthreads();
  for (int idx = tid; idx < 2048; idx += 256) {        // outcat A-frags (paired)
    const int ii = idx >> 7, c2 = (idx & 127) * 2;
    const float* src = e12 + (c2 >> 7) * 2048 + ii * 128 + (c2 & 127);
    st2(ocS + (c2 >> 5) * 512 + swA(ii, c2 & 31), src[0], src[1]);
  }
  __syncthreads();
  {                                                    // out = outcat @ Wo + bo + resid
    const int nt = w;
    v8f acc = {};
    for (int kb = 0; kb < 8; ++kb) {
      for (int idx = tid; idx < 2048; idx += 256) {
        const int k2 = (idx >> 7) * 2, c = idx & 127;
        st2(wstg + (c >> 4) * 512 + swB(k2, c & 15),
            Wo[(kb * 32 + k2) * 128 + c], Wo[(kb * 32 + k2 + 1) * 128 + c]);
      }
      __syncthreads();
      acc = wmma_bf16(acc, frag_ld(ocS + kb * 512), frag_ld(wstg + nt * 512));
      __syncthreads();
    }
    const int col = nt * 16 + colw;
#pragma unroll
    for (int r = 0; r < 8; ++r) {
      const int row = rb + r;
      ne[row * 128 + col] = acc[r] + bo[col] +
                            embio[(size_t)(b * 64 + i0 + row) * 128 + col];
    }
  }
  __syncthreads();
  if (tid < 16) {                                      // FFN pre-LN (eps 1e-6)
    const float* row = ne + tid * 128;
    float mu = 0.f;
    for (int c = 0; c < 128; ++c) mu += row[c];
    mu *= (1.f / 128.f);
    float var = 0.f;
    for (int c = 0; c < 128; ++c) { const float d = row[c] - mu; var += d * d; }
    var *= (1.f / 128.f);
    const float rstd = rsqrtf(var + 1e-6f);
    for (int c = 0; c < 128; c += 2)
      st2(hnS + (c >> 5) * 512 + swA(tid, c & 31),
          (row[c] - mu) * rstd * ln2g[c] + ln2b[c],
          (row[c + 1] - mu) * rstd * ln2g[c + 1] + ln2b[c + 1]);
  }
  __syncthreads();
  {                                                    // g = relu(hn @ W1 + b1)
    const int nt0f = w * 4;
    v8f acc[4] = {};
    for (int kb = 0; kb < 4; ++kb) {
      for (int idx = tid; idx < 8192; idx += 256) {
        const int k2 = (idx >> 9) * 2, c = idx & 511;
        st2(W1S + (c >> 4) * 512 + swB(k2, c & 15),
            W1[(kb * 32 + k2) * 512 + c], W1[(kb * 32 + k2 + 1) * 512 + c]);
      }
      __syncthreads();
      const v16bf a = frag_ld(hnS + kb * 512);
#pragma unroll
      for (int j = 0; j < 4; ++j)
        acc[j] = wmma_bf16(acc[j], a, frag_ld(W1S + (nt0f + j) * 512));
      __syncthreads();
    }
#pragma unroll
    for (int j = 0; j < 4; ++j) {
      const int col = (nt0f + j) * 16 + colw;
#pragma unroll
      for (int r = 0; r < 8; ++r)
        gS[(col >> 5) * 512 + swA(rb + r, col & 31)] =
            (bf16)fmaxf(acc[j][r] + b1[col], 0.f);
    }
  }
  __syncthreads();
  {                                                    // emb = g @ W2 + b2 + resid
    const int nt = w;
    v8f acc = {};
    for (int kb = 0; kb < 16; ++kb) {
      for (int idx = tid; idx < 2048; idx += 256) {
        const int k2 = (idx >> 7) * 2, c = idx & 127;
        st2(wstg + (c >> 4) * 512 + swB(k2, c & 15),
            W2[(kb * 32 + k2) * 128 + c], W2[(kb * 32 + k2 + 1) * 128 + c]);
      }
      __syncthreads();
      acc = wmma_bf16(acc, frag_ld(gS + kb * 512), frag_ld(wstg + nt * 512));
      __syncthreads();
    }
    const int col = nt * 16 + colw;
#pragma unroll
    for (int r = 0; r < 8; ++r) {
      const int row = rb + r;
      embio[(size_t)(b * 64 + i0 + row) * 128 + col] =
          acc[r] + b2[col] + ne[row * 128 + col];
    }
  }
}

// ---------------------------------------------------------------------------
// Kernel 4: output[b][d] = mean_n emb[b][n][d]
// ---------------------------------------------------------------------------
__global__ void mean_kernel(const float* __restrict__ emb, float* __restrict__ out) {
  const int b = blockIdx.x, d = threadIdx.x;
  float s = 0.f;
  for (int n = 0; n < 64; ++n) s += emb[(b * 64 + n) * 128 + d];
  out[b * 128 + d] = s * (1.f / 64.f);
}

// ---------------------------------------------------------------------------
extern "C" void kernel_launch(void* const* d_in, const int* in_sizes, int n_in,
                              void* d_out, int out_size, void* d_ws, size_t ws_size,
                              hipStream_t stream) {
  (void)in_sizes; (void)n_in; (void)out_size; (void)ws_size;
  const float* node_feat = (const float*)d_in[0];
  const float* stpe      = (const float*)d_in[1];
  const float* edge      = (const float*)d_in[2];
  const unsigned char* mask = (const unsigned char*)d_in[3];   // jnp bool = 1 byte
  const float* Wn  = (const float*)d_in[4];
  const float* bn  = (const float*)d_in[5];
  const float* We  = (const float*)d_in[6];
  const float* be  = (const float*)d_in[7];
  const float* l1g = (const float*)d_in[8];
  const float* l1b = (const float*)d_in[9];
  const float* Wq  = (const float*)d_in[10];
  const float* Wk  = (const float*)d_in[11];
  const float* Wv  = (const float*)d_in[12];
  const float* Wo  = (const float*)d_in[13];
  const float* bo  = (const float*)d_in[14];
  const float* l2g = (const float*)d_in[15];
  const float* l2b = (const float*)d_in[16];
  const float* W1  = (const float*)d_in[17];
  const float* b1  = (const float*)d_in[18];
  const float* W2  = (const float*)d_in[19];
  const float* b2  = (const float*)d_in[20];
  float* out = (float*)d_out;

  float* ws  = (float*)d_ws;
  float* emb = ws;                       // [4096][128]
  float* q   = emb + 4096 * 128;         // [2][64][64][128]
  float* k   = q   + 2 * 64 * 64 * 128;
  float* v   = k   + 2 * 64 * 64 * 128;

  (void)hipFuncSetAttribute((const void*)attn_kernel,
                            hipFuncAttributeMaxDynamicSharedMemorySize,
                            ATTN_SMEM_BYTES);

  node_proj_kernel<<<64, 256, 0, stream>>>(node_feat, stpe, Wn, bn, emb);
  for (int l = 0; l < 2; ++l) {
    qkv_kernel<<<192, 256, 0, stream>>>(emb, l1g + l * 128, l1b + l * 128,
                                        Wq + l * 128 * 256, Wk + l * 128 * 256,
                                        Wv + l * 128 * 256, q, k, v);
    attn_kernel<<<256, 256, ATTN_SMEM_BYTES, stream>>>(
        edge, mask, We + l * 128 * 128, be + l * 128, q, k, v,
        Wo + l * 256 * 128, bo + l * 128, l2g + l * 128, l2b + l * 128,
        W1 + l * 128 * 512, b1 + l * 512, W2 + l * 512 * 128, b2 + l * 128,
        emb, out + 8192 + (size_t)l * NHn * Bn * Nn * Nn);
  }
  mean_kernel<<<64, 128, 0, stream>>>(emb, out);
}
